// TransformerMoE_90125593739688
// MI455X (gfx1250) — matrix operand order repaired and checked by reading, not compiled, not run on hardware
//
#include <hip/hip_runtime.h>
#include <math.h>

// Problem constants (match reference)
#define Bb 4
#define Ss 1024
#define INd 768
#define Dd 512
#define Hh 8
#define Ff 2048
#define Ee 8
#define Ll 2
#define Cc 2
#define DHh 64
#define Tt 4096   // B*S

typedef __attribute__((ext_vector_type(16))) __bf16 v16bf;
typedef __attribute__((ext_vector_type(8)))  float  v8f;

// ---- CDNA5 async global->LDS copy (ASYNCcnt path), with safe fallback ------
#if defined(__HIP_DEVICE_COMPILE__) && \
    __has_builtin(__builtin_amdgcn_global_load_async_to_lds_b128) && \
    __has_builtin(__builtin_amdgcn_s_wait_asynccnt)
#define HAVE_ASYNC 1
#else
#define HAVE_ASYNC 0
#endif

#if HAVE_ASYNC
typedef int v4i_cp __attribute__((vector_size(16)));
typedef __attribute__((address_space(1))) v4i_cp* g_v4i_p;   // global src
typedef __attribute__((address_space(3))) v4i_cp* l_v4i_p;   // LDS dst
#endif

__device__ __forceinline__ void cp16_g2l(void* lds, const void* g) {
#if HAVE_ASYNC
  __builtin_amdgcn_global_load_async_to_lds_b128((g_v4i_p)(g), (l_v4i_p)(lds), 0, 0);
#else
  *reinterpret_cast<uint4*>(lds) = *reinterpret_cast<const uint4*>(g);
#endif
}

__device__ __forceinline__ void cp_async_wait() {
#if HAVE_ASYNC
  __builtin_amdgcn_s_wait_asynccnt(0);
#endif
}

__device__ __forceinline__ unsigned short f2bf(float f) {
  union { float f; unsigned int u; } c; c.f = f;
  unsigned int u = c.u;
  return (unsigned short)((u + 0x7FFFu + ((u >> 16) & 1u)) >> 16);  // RNE
}

__device__ __forceinline__ v8f zero_v8() {
  v8f z;
#pragma unroll
  for (int i = 0; i < 8; ++i) z[i] = 0.f;
  return z;
}

__device__ __forceinline__ v8f bwmma(const unsigned int a[8], const unsigned int b[8], v8f c) {
  union { unsigned int u[8]; v16bf v; } A, Bv;
#pragma unroll
  for (int i = 0; i < 8; ++i) { A.u[i] = a[i]; Bv.u[i] = b[i]; }
  return __builtin_amdgcn_wmma_f32_16x16x32_bf16(false, A.v, false, Bv.v, (short)0, c, false, false);
}

// A fragment 16x32 bf16 from LDS (row-major, stride ld in bf16 units, even).
// ISA layout: lane m=lane&15; lanes<16: K 0..7 in v0..3, 16..23 in v4..7; lanes>=16: +8.
__device__ __forceinline__ void frag_a(const unsigned short* s, int ld, int m0, int koff,
                                       int lane, unsigned int fr[8]) {
  const unsigned short* row = s + (size_t)(m0 + (lane & 15)) * ld + koff;
  const int h8 = ((lane >> 4) & 1) << 3;
#pragma unroll
  for (int v = 0; v < 8; ++v) {
    int k0 = ((v & 3) << 1) + ((v >> 2) << 4) + h8;
    fr[v] = *reinterpret_cast<const unsigned int*>(row + k0);
  }
}

// B fragment 32x16 bf16 from LDS staged K-transposed: sT[n][k].
// Layout (matches frag_a): col n=lane&15; lanes<16: K 0..7 in v0..3, 16..23 in v4..7; lanes>=16: +8.
__device__ __forceinline__ void frag_b(const unsigned short* sT, int ld, int n0, int koff,
                                       int lane, unsigned int fr[8]) {
  const unsigned short* row = sT + (size_t)(n0 + (lane & 15)) * ld + koff;
  const int h8 = ((lane >> 4) & 1) << 3;
#pragma unroll
  for (int v = 0; v < 8; ++v)
    fr[v] = *reinterpret_cast<const unsigned int*>(row + ((v & 3) << 1) + ((v >> 2) << 4) + h8);
}

// ---------------------------------------------------------------------------
// Generic WMMA GEMM: Out[M,N] = epi(A[M,K] @ W[K,N] + bias)
// Block 128 thr (4 waves), tile 64x64, K step 64 (8 WMMA per barrier pair).
// GATHER: A row p -> rowidx[p] (expert token gather), guarded by *cnt.
// PE: sinusoidal positional encoding. GELU: exact gelu.
// SCAT: scale by rowscale[p], write row rowdst[p] (slot*T+token).
// OBF16: store bf16 (feeds attention / next GEMM's A).
// ---------------------------------------------------------------------------
#define LKS 72  // padded LDS K-stride (bf16 units) for the 64-wide K slice

template<bool ABF16, bool GATHER, bool PE, bool GELU, bool SCAT, bool OBF16>
__global__ __launch_bounds__(128) void k_gemm(
    const void* __restrict__ Ap, const float* __restrict__ W,
    const float* __restrict__ bias, void* __restrict__ Outp,
    int N, int Kd,
    const int* __restrict__ rowidx, const int* __restrict__ cnt,
    const float* __restrict__ rowscale, const int* __restrict__ rowdst) {
  __shared__ unsigned short sA[64 * LKS];
  __shared__ unsigned short sB[64 * LKS];
  const int tid = threadIdx.x, lane = tid & 31, w = tid >> 5;
  const int wr = w >> 1, wc = w & 1;
  const int m0 = blockIdx.x * 64, n0 = blockIdx.y * 64;
  const int cv = (GATHER || SCAT) ? *cnt : 0x7fffffff;
  if ((GATHER || SCAT) && m0 >= cv) return;

  v8f acc[2][2];
#pragma unroll
  for (int i = 0; i < 2; ++i)
#pragma unroll
    for (int j = 0; j < 2; ++j) acc[i][j] = zero_v8();

  for (int kt = 0; kt < Kd; kt += 64) {
    // --- A tile: 64 rows x 64 k (bf16 in LDS) ---
    if (ABF16) {
#pragma unroll
      for (int it = 0; it < 4; ++it) {
        int i = tid + it * 128;                 // 512 chunks of 8 bf16 (16B)
        int row = i >> 3, c = (i & 7) << 3;
        int p = m0 + row;
        int g = GATHER ? ((p < cv) ? rowidx[p] : 0) : p;
        cp16_g2l(&sA[row * LKS + c],
                 (const unsigned short*)Ap + (size_t)g * Kd + kt + c);
      }
    } else {
#pragma unroll
      for (int it = 0; it < 8; ++it) {
        int i = tid + it * 128;                 // 1024 chunks of 4 floats
        int row = i >> 4, c = (i & 15) << 2;
        int p = m0 + row;
        int g = GATHER ? ((p < cv) ? rowidx[p] : 0) : p;
        const float4 f = *reinterpret_cast<const float4*>(
            (const float*)Ap + (size_t)g * Kd + kt + c);
        uint2 pk;
        pk.x = (unsigned int)f2bf(f.x) | ((unsigned int)f2bf(f.y) << 16);
        pk.y = (unsigned int)f2bf(f.z) | ((unsigned int)f2bf(f.w) << 16);
        *reinterpret_cast<uint2*>(&sA[row * LKS + c]) = pk;
      }
    }
    // --- B tile: stage transposed sB[n][k] ---
#pragma unroll
    for (int it = 0; it < 8; ++it) {
      int i = tid + it * 128;                   // 1024 chunks of 4 floats
      int k = i >> 4, nc = (i & 15) << 2;
      const float* wsrc = W + (size_t)(kt + k) * N + n0 + nc;
      const float4 f = *reinterpret_cast<const float4*>(wsrc);
      if ((i & 15) == 0 && kt + 64 < Kd)
        __builtin_prefetch(wsrc + (size_t)64 * N, 0, 1);  // next K tile
      sB[(nc + 0) * LKS + k] = f2bf(f.x);
      sB[(nc + 1) * LKS + k] = f2bf(f.y);
      sB[(nc + 2) * LKS + k] = f2bf(f.z);
      sB[(nc + 3) * LKS + k] = f2bf(f.w);
    }
    cp_async_wait();
    __syncthreads();
#pragma unroll
    for (int ks = 0; ks < 2; ++ks) {
      unsigned int fa[2][8], fb[2][8];
      frag_a(sA, LKS, wr * 32 + 0,  ks * 32, lane, fa[0]);
      frag_a(sA, LKS, wr * 32 + 16, ks * 32, lane, fa[1]);
      frag_b(sB, LKS, wc * 32 + 0,  ks * 32, lane, fb[0]);
      frag_b(sB, LKS, wc * 32 + 16, ks * 32, lane, fb[1]);
#pragma unroll
      for (int mi = 0; mi < 2; ++mi)
#pragma unroll
        for (int ni = 0; ni < 2; ++ni)
          acc[mi][ni] = bwmma(fa[mi], fb[ni], acc[mi][ni]);
    }
    __syncthreads();
  }

  // Epilogue. C layout: VGPR r -> row r (lanes<16) / r+8 (lanes>=16); col = lane&15.
  const int col = lane & 15, h8 = ((lane >> 4) & 1) << 3;
#pragma unroll
  for (int mi = 0; mi < 2; ++mi)
#pragma unroll
    for (int ni = 0; ni < 2; ++ni)
#pragma unroll
      for (int r = 0; r < 8; ++r) {
        int p = m0 + wr * 32 + mi * 16 + r + h8;
        int n = n0 + wc * 32 + ni * 16 + col;
        float v = acc[mi][ni][r];
        if (bias) v += bias[n];
        if (PE) {
          int s = p & (Ss - 1);
          float ex = (float)(n & ~1) * (-9.210340372f / (float)Dd);  // -ln(1e4)/D * 2j
          float ang = (float)s * __expf(ex);
          v += (n & 1) ? cosf(ang) : sinf(ang);
        }
        if (GELU) v = 0.5f * v * (1.f + erff(v * 0.70710678f));
        if (SCAT) {
          if (p < cv)
            ((float*)Outp)[(size_t)rowdst[p] * N + n] = v * rowscale[p];
        } else if (OBF16) {
          ((unsigned short*)Outp)[(size_t)p * N + n] = f2bf(v);
        } else {
          ((float*)Outp)[(size_t)p * N + n] = v;
        }
      }
}

// ---------------------------------------------------------------------------
// Fused flash attention: one block = (b,h) x 64 q-rows. 4 waves x 16 rows each.
// Q,K,V bf16 [T, H*DH]; online softmax; scores never touch HBM.
// ---------------------------------------------------------------------------
__global__ __launch_bounds__(128) void k_attn(
    const unsigned short* __restrict__ Q, const unsigned short* __restrict__ Km,
    const unsigned short* __restrict__ V, float* __restrict__ O) {
  __shared__ unsigned short sQ[64 * 72];
  __shared__ unsigned short sK[64 * 72];       // [key][dh] == B^T layout for Q@K^T
  __shared__ unsigned short sV[64 * 72];       // transposed: [dh][key] for P@V
  __shared__ unsigned short sP[4 * 16 * 72];   // per-wave P tiles (bf16)
  const int tid = threadIdx.x, lane = tid & 31, w = tid >> 5;
  const int col = lane & 15, h8 = ((lane >> 4) & 1) << 3;
  const int q0 = blockIdx.x * 64;
  const int b = blockIdx.y >> 3, h = blockIdx.y & 7;
  const size_t base = ((size_t)b * Ss) * Dd + (size_t)h * DHh;

#pragma unroll
  for (int it = 0; it < 4; ++it) {             // 512 chunks of 8 bf16 (16B)
    int i = tid + it * 128; int r = i >> 3, c = (i & 7) << 3;
    cp16_g2l(&sQ[r * 72 + c], &Q[base + (size_t)(q0 + r) * Dd + c]);
  }
  cp_async_wait();

  float mrun[8], lrun[8];
  v8f oacc[4];
#pragma unroll
  for (int r = 0; r < 8; ++r) { mrun[r] = -1e30f; lrun[r] = 0.f; }
#pragma unroll
  for (int ni = 0; ni < 4; ++ni) oacc[ni] = zero_v8();

  for (int kt = 0; kt < Ss / 64; ++kt) {
    __syncthreads();
    const int k0 = kt * 64;
#pragma unroll
    for (int it = 0; it < 4; ++it) {           // K tile: async 16B copies
      int i = tid + it * 128; int r = i >> 3, c = (i & 7) << 3;
      cp16_g2l(&sK[r * 72 + c], &Km[base + (size_t)(k0 + r) * Dd + c]);
    }
#pragma unroll
    for (int it = 0; it < 8; ++it) {           // V tile: transposed store
      int i = tid + it * 128; int r = i >> 4, c = (i & 15) << 2;
      uint2 vv = *reinterpret_cast<const uint2*>(&V[base + (size_t)(k0 + r) * Dd + c]);
      sV[(c + 0) * 72 + r] = (unsigned short)(vv.x & 0xffff);
      sV[(c + 1) * 72 + r] = (unsigned short)(vv.x >> 16);
      sV[(c + 2) * 72 + r] = (unsigned short)(vv.y & 0xffff);
      sV[(c + 3) * 72 + r] = (unsigned short)(vv.y >> 16);
    }
    cp_async_wait();
    __syncthreads();

    // S = Q @ K^T  (16 q-rows per wave x 64 keys)
    v8f sacc[4];
#pragma unroll
    for (int ni = 0; ni < 4; ++ni) sacc[ni] = zero_v8();
#pragma unroll
    for (int ks = 0; ks < 2; ++ks) {
      unsigned int fa[8]; frag_a(sQ, 72, w * 16, ks * 32, lane, fa);
#pragma unroll
      for (int ni = 0; ni < 4; ++ni) {
        unsigned int fb[8]; frag_b(sK, 72, ni * 16, ks * 32, lane, fb);
        sacc[ni] = bwmma(fa, fb, sacc[ni]);
      }
    }
#pragma unroll
    for (int ni = 0; ni < 4; ++ni)
#pragma unroll
      for (int r = 0; r < 8; ++r) sacc[ni][r] *= 0.125f;  // 1/sqrt(DH)

    // online softmax per row (row r+h8; 16 lanes of a half share a row)
#pragma unroll
    for (int r = 0; r < 8; ++r) {
      float mx = -1e30f;
#pragma unroll
      for (int ni = 0; ni < 4; ++ni) mx = fmaxf(mx, sacc[ni][r]);
#pragma unroll
      for (int mk = 1; mk < 16; mk <<= 1) mx = fmaxf(mx, __shfl_xor(mx, mk, 32));
      float mnew = fmaxf(mrun[r], mx);
      float alpha = __expf(mrun[r] - mnew);
      float rs = 0.f;
#pragma unroll
      for (int ni = 0; ni < 4; ++ni) {
        float pv = __expf(sacc[ni][r] - mnew); sacc[ni][r] = pv; rs += pv;
      }
#pragma unroll
      for (int mk = 1; mk < 16; mk <<= 1) rs += __shfl_xor(rs, mk, 32);
      lrun[r] = lrun[r] * alpha + rs;
#pragma unroll
      for (int ni = 0; ni < 4; ++ni) oacc[ni][r] *= alpha;
      mrun[r] = mnew;
    }

    // transpose P (C-layout) -> A-fragment layout through wave-private LDS
#pragma unroll
    for (int ni = 0; ni < 4; ++ni)
#pragma unroll
      for (int r = 0; r < 8; ++r)
        sP[(w * 16 + r + h8) * 72 + ni * 16 + col] = f2bf(sacc[ni][r]);
    asm volatile("s_wait_dscnt 0" ::: "memory");  // in-wave LDS RAW before frag read

    // O += P @ V
#pragma unroll
    for (int ks = 0; ks < 2; ++ks) {
      unsigned int fa[8]; frag_a(sP, 72, w * 16, ks * 32, lane, fa);
#pragma unroll
      for (int ni = 0; ni < 4; ++ni) {
        unsigned int fb[8]; frag_b(sV, 72, ni * 16, ks * 32, lane, fb);
        oacc[ni] = bwmma(fa, fb, oacc[ni]);
      }
    }
  }

#pragma unroll
  for (int ni = 0; ni < 4; ++ni)
#pragma unroll
    for (int r = 0; r < 8; ++r) {
      int qq = q0 + w * 16 + r + h8;
      int dh = ni * 16 + col;
      O[base + (size_t)qq * Dd + dh] = oacc[ni][r] / lrun[r];
    }
}

// ---------------------------------------------------------------------------
// MoE gate: softmax, top-2 (first-index ties), normalized weights,
// per-expert token lists via int atomics (value-deterministic).
// ---------------------------------------------------------------------------
__global__ void k_gate(const float* __restrict__ x, const float* __restrict__ gw,
                       const float* __restrict__ gb, float* __restrict__ gs,
                       int* __restrict__ counts, int* __restrict__ rtok,
                       int* __restrict__ rdst, float* __restrict__ rwt) {
  int t = blockIdx.x * blockDim.x + threadIdx.x;
  if (t >= Tt) return;
  float acc[Ee];
#pragma unroll
  for (int e = 0; e < Ee; ++e) acc[e] = gb[e];
  const float* xr = x + (size_t)t * Dd;
  for (int d = 0; d < Dd; ++d) {
    float xv = xr[d];
    const float* gr = gw + (size_t)d * Ee;
#pragma unroll
    for (int e = 0; e < Ee; ++e) acc[e] += xv * gr[e];
  }
  float mx = acc[0];
#pragma unroll
  for (int e = 1; e < Ee; ++e) mx = fmaxf(mx, acc[e]);
  float sum = 0.f;
#pragma unroll
  for (int e = 0; e < Ee; ++e) { acc[e] = __expf(acc[e] - mx); sum += acc[e]; }
  float inv = 1.f / sum;
#pragma unroll
  for (int e = 0; e < Ee; ++e) { acc[e] *= inv; gs[(size_t)t * Ee + e] = acc[e]; }
  int i0 = 0;
#pragma unroll
  for (int e = 1; e < Ee; ++e) if (acc[e] > acc[i0]) i0 = e;
  int i1 = (i0 == 0) ? 1 : 0;
#pragma unroll
  for (int e = 0; e < Ee; ++e) if (e != i0 && acc[e] > acc[i1]) i1 = e;
  float wn = acc[i0] + acc[i1];
  float w0 = acc[i0] / wn, w1 = acc[i1] / wn;
  int p0 = atomicAdd(&counts[i0], 1);
  rtok[i0 * Tt + p0] = t; rdst[i0 * Tt + p0] = t;      rwt[i0 * Tt + p0] = w0;  // slot0
  int p1 = atomicAdd(&counts[i1], 1);
  rtok[i1 * Tt + p1] = t; rdst[i1 * Tt + p1] = Tt + t; rwt[i1 * Tt + p1] = w1;  // slot1
}

__global__ void k_route(const float* __restrict__ gs, float* __restrict__ routing) {
  __shared__ float red[256];
  int e = blockIdx.x;
  float s = 0.f;
  for (int t = threadIdx.x; t < Tt; t += 256) s += gs[(size_t)t * Ee + e];
  red[threadIdx.x] = s; __syncthreads();
  for (int st = 128; st > 0; st >>= 1) {
    if (threadIdx.x < st) red[threadIdx.x] += red[threadIdx.x + st];
    __syncthreads();
  }
  if (threadIdx.x == 0) routing[e] = red[0];
}

__global__ void k_aux(const int* __restrict__ counts, const float* __restrict__ routing,
                      float* __restrict__ aux) {
  if (threadIdx.x == 0 && blockIdx.x == 0) {
    float a = 0.f;
    for (int e = 0; e < Ee; ++e)
      a += ((float)counts[e] / (float)(Tt * 2)) * (routing[e] / (float)Tt);
    aux[0] += (float)Ee * a;
  }
}

// x[t] = LN(x[t] + y[t] (+ y2[t])) * sc + bi   (one block per token)
__global__ __launch_bounds__(128) void k_ln(float* __restrict__ x, const float* __restrict__ y,
                                            const float* __restrict__ y2,
                                            const float* __restrict__ sc,
                                            const float* __restrict__ bi) {
  int t = blockIdx.x, tid = threadIdx.x, lane = tid & 31, w = tid >> 5;
  float v[4]; float s = 0.f, s2 = 0.f;
#pragma unroll
  for (int i = 0; i < 4; ++i) {
    int d = tid + i * 128;
    float a = x[(size_t)t * Dd + d] + y[(size_t)t * Dd + d];
    if (y2) a += y2[(size_t)t * Dd + d];
    v[i] = a; s += a; s2 += a * a;
  }
#pragma unroll
  for (int mk = 1; mk < 32; mk <<= 1) { s += __shfl_xor(s, mk, 32); s2 += __shfl_xor(s2, mk, 32); }
  __shared__ float rs[4], rs2[4];
  if (lane == 0) { rs[w] = s; rs2[w] = s2; }
  __syncthreads();
  s = rs[0] + rs[1] + rs[2] + rs[3];
  s2 = rs2[0] + rs2[1] + rs2[2] + rs2[3];
  float mu = s * (1.f / Dd);
  float var = s2 * (1.f / Dd) - mu * mu;
  float inv = rsqrtf(var + 1e-5f);
#pragma unroll
  for (int i = 0; i < 4; ++i) {
    int d = tid + i * 128;
    x[(size_t)t * Dd + d] = (v[i] - mu) * inv * sc[d] + bi[d];
  }
}

__global__ void k_pool(const float* __restrict__ h, float* __restrict__ pooled) {
  int i = blockIdx.x * blockDim.x + threadIdx.x;
  if (i >= Bb * Dd) return;
  int b = i >> 9, d = i & (Dd - 1);
  float s = 0.f;
  for (int t = 0; t < Ss; ++t) s += h[((size_t)b * Ss + t) * Dd + d];
  pooled[i] = s * (1.f / Ss);
}

__global__ void k_head(const float* __restrict__ pooled, const float* __restrict__ cw,
                       const float* __restrict__ cb, const float* __restrict__ aux,
                       float* __restrict__ out) {
  int i = threadIdx.x;
  if (i < Bb * Cc) {
    int b = i >> 1, c = i & 1;
    float s = cb[c];
    for (int d = 0; d < Dd; ++d) s += pooled[b * Dd + d] * cw[d * Cc + c];
    out[i] = s;
  } else if (i == Bb * Cc) {
    out[Bb * Cc] = 0.01f * aux[0];  // LBW * total_aux
  }
}

// ---------------------------------------------------------------------------
extern "C" void kernel_launch(void* const* d_in, const int* in_sizes, int n_in,
                              void* d_out, int out_size, void* d_ws, size_t ws_size,
                              hipStream_t stream) {
  (void)in_sizes; (void)n_in; (void)out_size; (void)ws_size;
  const float* x      = (const float*)d_in[0];
  const float* proj_w = (const float*)d_in[1];
  const float* proj_b = (const float*)d_in[2];
  const float* attn_w = (const float*)d_in[3];
  const float* attn_b = (const float*)d_in[4];
  const float* ln1_s  = (const float*)d_in[5];
  const float* ln1_b  = (const float*)d_in[6];
  const float* gate_w = (const float*)d_in[7];
  const float* gate_b = (const float*)d_in[8];
  const float* w1     = (const float*)d_in[9];
  const float* b1     = (const float*)d_in[10];
  const float* w2     = (const float*)d_in[11];
  const float* b2     = (const float*)d_in[12];
  const float* ln2_s  = (const float*)d_in[13];
  const float* ln2_b  = (const float*)d_in[14];
  const float* cls_w  = (const float*)d_in[15];
  const float* cls_b  = (const float*)d_in[16];
  float* out = (float*)d_out;

  char* wp = (char*)d_ws;
  auto alloc = [&](size_t bytes) -> void* {
    void* p = wp; wp += (bytes + 255) & ~(size_t)255; return p;
  };
  float*          hbuf   = (float*)alloc((size_t)Tt * Dd * 4);
  float*          attnb  = (float*)alloc((size_t)Tt * Dd * 4);
  unsigned short* qb     = (unsigned short*)alloc((size_t)Tt * Dd * 2);
  unsigned short* kb     = (unsigned short*)alloc((size_t)Tt * Dd * 2);
  unsigned short* vb     = (unsigned short*)alloc((size_t)Tt * Dd * 2);
  unsigned short* Hg     = (unsigned short*)alloc((size_t)Tt * Ff * 2);
  float*          yk     = (float*)alloc((size_t)2 * Tt * Dd * 4);  // also out-proj tmp
  float*          gs     = (float*)alloc((size_t)Tt * Ee * 4);
  int*            rtok   = (int*)alloc((size_t)Ee * Tt * 4);
  int*            rdst   = (int*)alloc((size_t)Ee * Tt * 4);
  float*          rwt    = (float*)alloc((size_t)Ee * Tt * 4);
  int*            counts = (int*)alloc(Ee * 4);
  float*          routing= (float*)alloc(Ee * 4);
  float*          aux    = (float*)alloc(4);
  float*          pooled = (float*)alloc((size_t)Bb * Dd * 4);

  (void)hipMemsetAsync(aux, 0, 4, stream);
  dim3 blk(128);

  // h = x @ proj_w + proj_b + PE   [4096,768]x[768,512]
  k_gemm<false,false,true,false,false,false><<<dim3(64, 8), blk, 0, stream>>>(
      x, proj_w, proj_b, hbuf, Dd, INd, nullptr, nullptr, nullptr, nullptr);

  for (int l = 0; l < Ll; ++l) {
    const float* aw = attn_w + (size_t)l * 4 * Dd * Dd;
    const float* ab = attn_b + (size_t)l * 4 * Dd;
    // Q, K, V (bf16 out -> feeds flash attention directly)
    k_gemm<false,false,false,false,false,true><<<dim3(64, 8), blk, 0, stream>>>(
        hbuf, aw + 0 * Dd * Dd, ab + 0 * Dd, qb, Dd, Dd, nullptr, nullptr, nullptr, nullptr);
    k_gemm<false,false,false,false,false,true><<<dim3(64, 8), blk, 0, stream>>>(
        hbuf, aw + 1 * Dd * Dd, ab + 1 * Dd, kb, Dd, Dd, nullptr, nullptr, nullptr, nullptr);
    k_gemm<false,false,false,false,false,true><<<dim3(64, 8), blk, 0, stream>>>(
        hbuf, aw + 2 * Dd * Dd, ab + 2 * Dd, vb, Dd, Dd, nullptr, nullptr, nullptr, nullptr);
    // fused flash attention
    k_attn<<<dim3(Ss / 64, Bb * Hh), blk, 0, stream>>>(qb, kb, vb, attnb);
    // out-proj -> yk (tmp), then LN1 in place
    k_gemm<false,false,false,false,false,false><<<dim3(64, 8), blk, 0, stream>>>(
        attnb, aw + 3 * Dd * Dd, ab + 3 * Dd, yk, Dd, Dd, nullptr, nullptr, nullptr, nullptr);
    k_ln<<<Tt, blk, 0, stream>>>(hbuf, yk, nullptr, ln1_s + l * Dd, ln1_b + l * Dd);

    // MoE: gate + top-2 routing
    (void)hipMemsetAsync(counts, 0, Ee * 4, stream);
    k_gate<<<Tt / 128, blk, 0, stream>>>(hbuf, gate_w + (size_t)l * Dd * Ee, gate_b + l * Ee,
                                         gs, counts, rtok, rdst, rwt);
    k_route<<<Ee, 256, 0, stream>>>(gs, routing);
    k_aux<<<1, 32, 0, stream>>>(counts, routing, aux);

    // per-expert gathered FFN (grid covers worst case, device-side count early-exit)
    for (int e = 0; e < Ee; ++e) {
      k_gemm<false,true,false,true,false,true><<<dim3(64, Ff / 64), blk, 0, stream>>>(
          hbuf, w1 + ((size_t)l * Ee + e) * Dd * Ff, b1 + ((size_t)l * Ee + e) * Ff,
          Hg, Ff, Dd, rtok + e * Tt, counts + e, nullptr, nullptr);
      k_gemm<true,false,false,false,true,false><<<dim3(64, 8), blk, 0, stream>>>(
          Hg, w2 + ((size_t)l * Ee + e) * Ff * Dd, b2 + ((size_t)l * Ee + e) * Dd,
          yk, Dd, Ff, nullptr, counts + e, rwt + e * Tt, rdst + e * Tt);
    }
    // LN2: h = LN(h + yk_slot0 + yk_slot1)
    k_ln<<<Tt, blk, 0, stream>>>(hbuf, yk, yk + (size_t)Tt * Dd, ln2_s + l * Dd, ln2_b + l * Dd);
  }

  k_pool<<<(Bb * Dd + 127) / 128, blk, 0, stream>>>(hbuf, pooled);
  k_head<<<1, 32, 0, stream>>>(pooled, cls_w, cls_b, aux, out);
}
